// TriangleAttention_12455405158978
// MI455X (gfx1250) — matrix operand order repaired and checked by reading, not compiled, not run on hardware
//
#include <hip/hip_runtime.h>
#include <hip/hip_bf16.h>

typedef __attribute__((ext_vector_type(16))) _Float16 v16h;
typedef __attribute__((ext_vector_type(8)))  _Float16 v8h;
typedef __attribute__((ext_vector_type(4)))  _Float16 v4h;
typedef __attribute__((ext_vector_type(8)))  float    v8f;
typedef __attribute__((ext_vector_type(4)))  float    v4f;

#define WMMA_F16(A,B,C) __builtin_amdgcn_wmma_f32_16x16x32_f16(false,(A),false,(B),(short)0,(C),false,false)

// Assemble a 16x32 f16 A-fragment for one lane from two contiguous 8-half groups.
// Lanes 0-15 row M=lane, K = {0..15}; lanes 16-31 row M=lane-16, K = {16..31}.
__device__ inline v16h make_a(const _Float16* p0, const _Float16* p1) {
    v8h lo = *(const v8h*)p0;
    v8h hi = *(const v8h*)p1;
    v16h r;
#pragma unroll
    for (int e = 0; e < 8; ++e) { r[e] = lo[e]; r[e + 8] = hi[e]; }
    return r;
}

// ---------------- Stage 0: fp32 -> fp16 weight conversion ----------------
__global__ void ta_cvt_f2h(const float* __restrict__ src, _Float16* __restrict__ dst, int n) {
    int t = blockIdx.x * 256 + threadIdx.x;
    if (t < n) dst[t] = (_Float16)src[t];
}

// ---------------- Stage 1: LayerNorm over C=128, one wave per (i,j) ----------------
__global__ __launch_bounds__(256) void ta_layernorm(
    const float* __restrict__ x, const float* __restrict__ lnw,
    const float* __restrict__ lnb, _Float16* __restrict__ xnh) {
    int wid = threadIdx.x >> 5, lane = threadIdx.x & 31;
    size_t p = (size_t)blockIdx.x * 8 + wid;          // position index, < 65536
    v4f v = *(const v4f*)(x + p * 128 + lane * 4);
    float s  = v[0] + v[1] + v[2] + v[3];
    float sq = v[0]*v[0] + v[1]*v[1] + v[2]*v[2] + v[3]*v[3];
#pragma unroll
    for (int off = 16; off >= 1; off >>= 1) {
        s  += __shfl_xor(s,  off, 32);
        sq += __shfl_xor(sq, off, 32);
    }
    float mu  = s * (1.0f / 128.0f);
    float var = sq * (1.0f / 128.0f) - mu * mu;
    float rs  = rsqrtf(var + 1e-5f);
    int c0 = lane * 4;
    v4h out;
#pragma unroll
    for (int t = 0; t < 4; ++t)
        out[t] = (_Float16)((v[t] - mu) * rs * lnw[c0 + t] + lnb[c0 + t]);
    *(v4h*)(xnh + p * 128 + c0) = out;
}

// ---------------- Stage 2: projections (Q,K,V,gate) + triangle bias ----------------
__global__ __launch_bounds__(128) void ta_proj(
    const _Float16* __restrict__ xn,
    const _Float16* __restrict__ wq, const _Float16* __restrict__ wk,
    const _Float16* __restrict__ wv, const _Float16* __restrict__ wg,
    const _Float16* __restrict__ triw, const float* __restrict__ wgb,
    _Float16* __restrict__ qout, _Float16* __restrict__ kout,
    _Float16* __restrict__ vtout, _Float16* __restrict__ gout,
    float* __restrict__ triout) {
    int wid = threadIdx.x >> 5, lane = threadIdx.x & 31;
    int rt = blockIdx.x * 4 + wid;                    // row tile 0..4095
    int m = lane & 15, hi = lane >> 4;
    int i  = rt >> 4;                                 // outer row index (constant per tile)
    int j0 = ((rt & 15) << 4) + hi * 8;               // inner col base; j = j0 + r

    const _Float16* xrow = xn + (size_t)(rt * 16 + m) * 128;
    v16h a[4];
#pragma unroll
    for (int kt = 0; kt < 4; ++kt) {
        int d0 = kt * 32 + hi * 16;
        a[kt] = make_a(xrow + d0, xrow + d0 + 8);
    }

    const _Float16* Ws[4] = {wq, wk, wv, wg};
#pragma unroll
    for (int w = 0; w < 4; ++w) {
        const _Float16* W = Ws[w];
#pragma unroll
        for (int ct = 0; ct < 8; ++ct) {
            int n = ct * 16 + m;                      // output channel e = h*32+d
            const _Float16* wp = W + (size_t)n * 128 + hi * 16;
            v16h b[4];
#pragma unroll
            for (int kt = 0; kt < 4; ++kt) b[kt] = *(const v16h*)(wp + kt * 32);
            v8f acc = {};
#pragma unroll
            for (int kt = 0; kt < 4; ++kt) acc = WMMA_F16(a[kt], b[kt], acc);

            int h = n >> 5, d = n & 31;
            // convert all 8 results into distinct registers first, then burst the stores
            _Float16 tmp[8];
            if (w == 0) {
#pragma unroll
                for (int r = 0; r < 8; ++r)
                    tmp[r] = (_Float16)(acc[r] * 0.17677669529663689f); // 1/sqrt(32)
                _Float16* qp = qout + ((size_t)(i * 4 + h) * 256 + j0) * 32 + d;
#pragma unroll
                for (int r = 0; r < 8; ++r) qp[r * 32] = tmp[r];
            } else if (w == 1) {
#pragma unroll
                for (int r = 0; r < 8; ++r) tmp[r] = (_Float16)acc[r];
                _Float16* kp = kout + ((size_t)(i * 4 + h) * 256 + j0) * 32 + d;
#pragma unroll
                for (int r = 0; r < 8; ++r) kp[r * 32] = tmp[r];
            } else if (w == 2) {
#pragma unroll
                for (int r = 0; r < 8; ++r) tmp[r] = (_Float16)acc[r];
                _Float16* vp = vtout + (size_t)(i * 4 + h) * 8192 + (size_t)d * 256 + j0;
#pragma unroll
                for (int r = 0; r < 8; ++r) vp[r] = tmp[r];
            } else {
                float bias = wgb[n];
#pragma unroll
                for (int r = 0; r < 8; ++r)
                    tmp[r] = (_Float16)(1.0f / (1.0f + __expf(-(acc[r] + bias))));
                _Float16* gp = gout + (size_t)(rt * 16 + hi * 8) * 128 + n;
#pragma unroll
                for (int r = 0; r < 8; ++r) gp[r * 128] = tmp[r];
            }
        }
    }
    // triangle bias: 4 head columns, padded into one 16-wide tile
    {
        v16h bz;
#pragma unroll
        for (int e = 0; e < 16; ++e) bz[e] = (_Float16)0.0f;
        v8f acc = {};
#pragma unroll
        for (int kt = 0; kt < 4; ++kt) {
            v16h b = (m < 4) ? *(const v16h*)(triw + m * 128 + kt * 32 + hi * 16) : bz;
            acc = WMMA_F16(a[kt], b, acc);
        }
        if (m < 4) {
            float* tp = triout + (size_t)m * 65536 + rt * 16 + hi * 8;
#pragma unroll
            for (int r = 0; r < 8; ++r) tp[r] = acc[r];
        }
    }
}

// ---------------- Stage 3: attention, 4 waves/block, scores in registers ----------------
__global__ __launch_bounds__(128) void ta_attn(
    const _Float16* __restrict__ qh, const _Float16* __restrict__ kh,
    const _Float16* __restrict__ vt, const _Float16* __restrict__ gh,
    const float* __restrict__ tri, const float* __restrict__ mask,
    _Float16* __restrict__ oh) {
    __shared__ alignas(16) _Float16 p_lds_all[4][16 * 264];
    int wid = threadIdx.x >> 5, lane = threadIdx.x & 31;
    int b = blockIdx.x;                               // = ih*4 + qgroup
    int ih = b >> 2;                                  // i*4 + h
    int h = ih & 3, i = ih >> 2;
    int qt = (b & 3) * 4 + wid;                       // query tile 0..15
    int q0 = qt * 16;
    int m = lane & 15, hi = lane >> 4;
    _Float16* p_lds = p_lds_all[wid];

    // Q fragment (K-dim == D == 32 exactly)
    const _Float16* qrow = qh + ((size_t)ih * 256 + (q0 + m)) * 32;
    v16h aq = make_a(qrow + hi * 16, qrow + hi * 16 + 8);

    // per-lane bias bases (C-fragment layout: col = kt*16+m, row = q0+hi*8+r)
    const float* trib  = tri + ((size_t)h * 256 + (q0 + hi * 8)) * 256 + m;
    const float* maskb = mask + i * 256 + m;
    const _Float16* kbase = kh + (size_t)ih * 256 * 32 + hi * 16;

    // ---- phase A: scores = Q.K^T with bias preloaded into C operand ----
    v8f sc[16];
    v16h bk = *(const v16h*)(kbase + (size_t)m * 32);
#pragma unroll
    for (int kt = 0; kt < 16; ++kt) {
        v16h bcur = bk;
        if (kt < 15) bk = *(const v16h*)(kbase + (size_t)((kt + 1) * 16 + m) * 32);
        float mb = 1e9f * (maskb[kt * 16] - 1.0f);
        v8f acc;
#pragma unroll
        for (int r = 0; r < 8; ++r) acc[r] = trib[r * 256 + kt * 16] + mb;
        sc[kt] = WMMA_F16(aq, bcur, acc);
    }

    // ---- phase B: softmax in registers (rows live in one VGPR across 16 lanes) ----
    float rmax[8], rinv[8];
#pragma unroll
    for (int r = 0; r < 8; ++r) {
        float mx = sc[0][r];
#pragma unroll
        for (int kt = 1; kt < 16; ++kt) mx = fmaxf(mx, sc[kt][r]);
#pragma unroll
        for (int off = 8; off >= 1; off >>= 1) mx = fmaxf(mx, __shfl_xor(mx, off, 32));
        rmax[r] = mx;
    }
#pragma unroll
    for (int kt = 0; kt < 16; ++kt)
#pragma unroll
        for (int r = 0; r < 8; ++r) sc[kt][r] = __expf(sc[kt][r] - rmax[r]);
#pragma unroll
    for (int r = 0; r < 8; ++r) {
        float s = 0.0f;
#pragma unroll
        for (int kt = 0; kt < 16; ++kt) s += sc[kt][r];
#pragma unroll
        for (int off = 8; off >= 1; off >>= 1) s += __shfl_xor(s, off, 32);
        rinv[r] = 1.0f / s;
    }
    // transpose probs through LDS (C-layout -> A-layout), fp16
#pragma unroll
    for (int kt = 0; kt < 16; ++kt)
#pragma unroll
        for (int r = 0; r < 8; ++r)
            p_lds[(r + hi * 8) * 264 + kt * 16 + m] = (_Float16)(sc[kt][r] * rinv[r]);
    __syncthreads();

    // ---- phase C: O = P.V, double-buffered V fragments ----
    const _Float16* vbase = vt + (size_t)ih * 8192 + hi * 16;
#pragma unroll
    for (int dt = 0; dt < 2; ++dt) {
        int d = dt * 16 + m;
        v8f acc = {};
        v16h bv = *(const v16h*)(vbase + (size_t)d * 256);
#pragma unroll
        for (int kc8 = 0; kc8 < 8; ++kc8) {
            v16h bcur = bv;
            if (kc8 < 7) bv = *(const v16h*)(vbase + (size_t)d * 256 + (kc8 + 1) * 32);
            int kk0 = kc8 * 32 + hi * 16;
            v16h ap = make_a(&p_lds[m * 264 + kk0], &p_lds[m * 264 + kk0 + 8]);
            acc = WMMA_F16(ap, bcur, acc);
        }
        // ---- phase D: gate + store, distinct-register conversion then store burst ----
        int e = h * 32 + d;
        const _Float16* gp = gh + ((size_t)i * 256 + q0 + hi * 8) * 128 + e;
        _Float16*       op = oh + ((size_t)i * 256 + q0 + hi * 8) * 128 + e;
        _Float16 tmp[8];
#pragma unroll
        for (int r = 0; r < 8; ++r)
            tmp[r] = (_Float16)(acc[r] * (float)gp[r * 128]);
#pragma unroll
        for (int r = 0; r < 8; ++r) op[r * 128] = tmp[r];
    }
}

// ---------------- Stage 4: output projection + bias ----------------
__global__ __launch_bounds__(128) void ta_outproj(
    const _Float16* __restrict__ oh, const _Float16* __restrict__ wo,
    const float* __restrict__ wob, float* __restrict__ out) {
    int wid = threadIdx.x >> 5, lane = threadIdx.x & 31;
    int rt = blockIdx.x * 4 + wid;
    int m = lane & 15, hi = lane >> 4;
    const _Float16* orow = oh + (size_t)(rt * 16 + m) * 128;
    v16h a[4];
#pragma unroll
    for (int kt = 0; kt < 4; ++kt) {
        int d0 = kt * 32 + hi * 16;
        a[kt] = make_a(orow + d0, orow + d0 + 8);
    }
#pragma unroll
    for (int ct = 0; ct < 8; ++ct) {
        int c = ct * 16 + m;
        const _Float16* wp = wo + (size_t)c * 128 + hi * 16;
        v16h b[4];
#pragma unroll
        for (int kt = 0; kt < 4; ++kt) b[kt] = *(const v16h*)(wp + kt * 32);
        v8f acc = {};
#pragma unroll
        for (int kt = 0; kt < 4; ++kt) acc = WMMA_F16(a[kt], b[kt], acc);
        float bias = wob[c];
        float* op = out + (size_t)(rt * 16 + hi * 8) * 128 + c;
#pragma unroll
        for (int r = 0; r < 8; ++r) op[r * 128] = acc[r] + bias;
    }
}

extern "C" void kernel_launch(void* const* d_in, const int* in_sizes, int n_in,
                              void* d_out, int out_size, void* d_ws, size_t ws_size,
                              hipStream_t stream) {
    const float* x    = (const float*)d_in[0];
    const float* mask = (const float*)d_in[1];
    const float* lnw  = (const float*)d_in[2];
    const float* lnb  = (const float*)d_in[3];
    const float* triw = (const float*)d_in[4];
    const float* wq   = (const float*)d_in[5];
    const float* wk   = (const float*)d_in[6];
    const float* wv   = (const float*)d_in[7];
    const float* wgw  = (const float*)d_in[8];
    const float* wgb  = (const float*)d_in[9];
    const float* wow  = (const float*)d_in[10];
    const float* wob  = (const float*)d_in[11];
    float* out = (float*)d_out;

    char* ws = (char*)d_ws;
    const size_t SZ_BIG = 16777216;    // 65536*128 halves (bytes)
    size_t o_xn  = 0;
    size_t o_q   = o_xn + SZ_BIG;
    size_t o_k   = o_q  + SZ_BIG;
    size_t o_v   = o_k  + SZ_BIG;
    size_t o_g   = o_v  + SZ_BIG;
    size_t o_o   = o_g  + SZ_BIG;
    size_t o_tri = o_o  + SZ_BIG;      // 4*65536 floats = 1 MB
    size_t o_wq  = o_tri + 1048576;
    size_t o_wk  = o_wq + 32768;
    size_t o_wv  = o_wk + 32768;
    size_t o_wg  = o_wv + 32768;
    size_t o_wo  = o_wg + 32768;
    size_t o_tw  = o_wo + 32768;       // tri_w fp16: 1 KB

    _Float16* xnh  = (_Float16*)(ws + o_xn);
    _Float16* qh   = (_Float16*)(ws + o_q);
    _Float16* kh   = (_Float16*)(ws + o_k);
    _Float16* vth  = (_Float16*)(ws + o_v);
    _Float16* gh   = (_Float16*)(ws + o_g);
    _Float16* ohb  = (_Float16*)(ws + o_o);
    float*    trib = (float*)(ws + o_tri);
    _Float16* wqh  = (_Float16*)(ws + o_wq);
    _Float16* wkh  = (_Float16*)(ws + o_wk);
    _Float16* wvh  = (_Float16*)(ws + o_wv);
    _Float16* wgh  = (_Float16*)(ws + o_wg);
    _Float16* woh  = (_Float16*)(ws + o_wo);
    _Float16* twh  = (_Float16*)(ws + o_tw);

    // Stage 0: weight conversion
    ta_cvt_f2h<<<64, 256, 0, stream>>>(wq,  wqh, 16384);
    ta_cvt_f2h<<<64, 256, 0, stream>>>(wk,  wkh, 16384);
    ta_cvt_f2h<<<64, 256, 0, stream>>>(wv,  wvh, 16384);
    ta_cvt_f2h<<<64, 256, 0, stream>>>(wgw, wgh, 16384);
    ta_cvt_f2h<<<64, 256, 0, stream>>>(wow, woh, 16384);
    ta_cvt_f2h<<<2,  256, 0, stream>>>(triw, twh, 512);

    // Stage 1: layernorm  (65536 positions, 8 waves/block)
    ta_layernorm<<<8192, 256, 0, stream>>>(x, lnw, lnb, xnh);

    // Stage 2: projections + triangle bias (4096 row tiles, 4 waves/block)
    ta_proj<<<1024, 128, 0, stream>>>(xnh, wqh, wkh, wvh, wgh, twh, wgb,
                                      qh, kh, vth, gh, trib);

    // Stage 3: attention (1024 (i,h) pairs * 4 query-tile groups, 4 waves/block)
    ta_attn<<<4096, 128, 0, stream>>>(qh, kh, vth, gh, trib, mask, ohb);

    // Stage 4: output projection
    ta_outproj<<<1024, 128, 0, stream>>>(ohb, woh, wob, out);
}
